// TransformerEncoderLayer_1941325218171
// MI455X (gfx1250) — matrix operand order repaired and checked by reading, not compiled, not run on hardware
//
#include <hip/hip_runtime.h>

// ---------------------------------------------------------------------------
// Types for CDNA5 WMMA (wave32): D(f32 16x16) = A(bf16 16x32) * B(bf16 32x16) + C
// ---------------------------------------------------------------------------
typedef __bf16 bf16_t;
typedef __attribute__((ext_vector_type(16))) __bf16 bf16x16;
typedef __attribute__((ext_vector_type(8)))  __bf16 bf16x8;
typedef __attribute__((ext_vector_type(8)))  float  f32x8;

// Problem sizes (fixed by reference)
#define MM 8192      // BATCH * SEQ
#define DD 1024      // d_model
#define FF 4096      // d_ff
#define SS 2048      // seq
#define DK 64        // head dim

// ---------------------------------------------------------------------------
// Fragment loaders (contiguous k order: lane half h holds k = 16h..16h+15 for A and B).
// ---------------------------------------------------------------------------
__device__ inline bf16x16 load_frag_a(const bf16_t* base, int stride, int lane) {
  const int m  = lane & 15;
  const int kb = (lane & 16) ? 16 : 0;
  const bf16_t* p = base + (size_t)m * stride + kb;
  union { bf16x16 v; struct { bf16x8 lo, hi; } s; } u;
  u.s.lo = *(const bf16x8*)(p);
  u.s.hi = *(const bf16x8*)(p + 8);
  return u.v;
}

__device__ inline bf16x16 load_frag_bT(const bf16_t* base, int stride, int lane) {
  const int n  = lane & 15;
  const int kb = (lane & 16) ? 16 : 0;
  const bf16_t* p = base + (size_t)n * stride + kb;
  union { bf16x16 v; struct { bf16x8 lo, hi; } s; } u;
  u.s.lo = *(const bf16x8*)(p);
  u.s.hi = *(const bf16x8*)(p + 8);
  return u.v;
}

__device__ inline f32x8 wmma_bf16(bf16x16 a, bf16x16 b, f32x8 c) {
  return __builtin_amdgcn_wmma_f32_16x16x32_bf16(false, a, false, b,
                                                 (short)0, c, false, false);
}

// CDNA5 async copy: global -> LDS, tracked by ASYNCcnt (ISA 15.18.3 op 98).
__device__ inline void async_ld_b128(unsigned lds_off, const void* gptr) {
  asm volatile("global_load_async_to_lds_b128 %0, %1, off"
               :: "v"(lds_off), "v"(gptr) : "memory");
}
__device__ inline unsigned lds_addr(const void* p) {
  return (unsigned)(size_t)p;   // low 32 bits of generic addr = LDS offset
}

// ---------------------------------------------------------------------------
// fp32 -> bf16 conversion (x only)
// ---------------------------------------------------------------------------
__global__ __launch_bounds__(256) void cvt_kernel(const float* __restrict__ in,
                                                  bf16_t* __restrict__ out, int n) {
  int i = blockIdx.x * 256 + threadIdx.x;
  if (i < n) out[i] = (bf16_t)in[i];
}

// ---------------------------------------------------------------------------
// fp32 [K][N] -> bf16 transposed [N][K] (weights; one-time).
// ---------------------------------------------------------------------------
__global__ __launch_bounds__(256) void cvtT_kernel(const float* __restrict__ in,
                                                   bf16_t* __restrict__ out,
                                                   int K, int N) {
  __shared__ bf16_t t[32][33];
  const int k0 = blockIdx.y * 32, n0 = blockIdx.x * 32;
  const int c = threadIdx.x & 31, r8 = threadIdx.x >> 5;
#pragma unroll
  for (int i = 0; i < 4; ++i) {
    int r = r8 + i * 8;
    t[r][c] = (bf16_t)in[(size_t)(k0 + r) * N + n0 + c];
  }
  __syncthreads();
#pragma unroll
  for (int i = 0; i < 4; ++i) {
    int r = r8 + i * 8;
    out[(size_t)(n0 + r) * K + k0 + c] = t[c][r];
  }
}

// ---------------------------------------------------------------------------
// Tiled bf16 WMMA GEMM with DOUBLE-BUFFERED ASYNC-to-LDS pipeline:
// tiles stream straight into LDS via global_load_async_to_lds_b128 (no VGPR
// round trip); next tile's copies are issued before computing the current one
// (async loads complete in order -> s_wait_asynccnt 0x3 covers the current 3).
// out = A[MxK] * W[KxN] + bias, W given TRANSPOSED (BwT[N][K]). MODE:
//   0: store bf16 row-major, scaled by `sc`
//   1: store f32 (acc + bias + Rres)      (residual add)
//   2: exact GELU, store bf16 row-major
//   3: store bf16 TRANSPOSED [N][M]       (for V^T; fully vectorized)
// ---------------------------------------------------------------------------
template <int MODE>
__global__ __launch_bounds__(256) void gemm_bf16_wmma(
    const bf16_t* __restrict__ A, const bf16_t* __restrict__ BwT,
    const float* __restrict__ bias, const float* __restrict__ Rres,
    void* __restrict__ outp, int M, int N, int K, float sc) {
  __shared__ bf16_t As[2][64][32];       // row-major [m][k]
  __shared__ bf16_t Bs[2][128][40];      // K-major per column [n][k], pad 40

  const int tid  = threadIdx.x;
  const int lane = tid & 31;
  const int w    = tid >> 5;
  const int wc   = w & 3;
  const int wr   = w >> 2;
  const int m0   = blockIdx.y * 64;
  const int n0   = blockIdx.x * 128;

  // per-thread copy chunks (all 16B)
  const int ar  = tid >> 2, ac8 = (tid & 3) * 8;       // A: 64x32
  const int bn0 = tid >> 2, bc8 = (tid & 3) * 8;       // B: rows bn0, bn0+64
  const bf16_t* Ap  = A   + (size_t)(m0 + ar) * K + ac8;
  const bf16_t* Bp0 = BwT + (size_t)(n0 + bn0) * K + bc8;
  const bf16_t* Bp1 = BwT + (size_t)(n0 + bn0 + 64) * K + bc8;

  f32x8 acc[2][2] = {};

  // issue async copies of tile at k0 into buffer `buf` (3 per thread)
  auto issue = [&](int k0, int buf) {
    async_ld_b128(lds_addr(&As[buf][ar][ac8]),        Ap  + k0);
    async_ld_b128(lds_addr(&Bs[buf][bn0][bc8]),       Bp0 + k0);
    async_ld_b128(lds_addr(&Bs[buf][bn0 + 64][bc8]),  Bp1 + k0);
  };

  issue(0, 0);
  for (int k0 = 0; k0 < K; k0 += 32) {
    const int  cur  = (k0 >> 5) & 1;
    const bool more = (k0 + 32 < K);
    if (more) {
      issue(k0 + 32, cur ^ 1);
      asm volatile("s_wait_asynccnt 0x3" ::: "memory");  // current 3 done
    } else {
      asm volatile("s_wait_asynccnt 0x0" ::: "memory");
    }
    __syncthreads();                                     // all waves' copies in

    bf16x16 af[2], bfr[2];
    af[0]  = load_frag_a (&As[cur][wr * 32 +  0][0], 32, lane);
    af[1]  = load_frag_a (&As[cur][wr * 32 + 16][0], 32, lane);
    bfr[0] = load_frag_bT(&Bs[cur][wc * 32 +  0][0], 40, lane);
    bfr[1] = load_frag_bT(&Bs[cur][wc * 32 + 16][0], 40, lane);
#pragma unroll
    for (int mi = 0; mi < 2; ++mi)
#pragma unroll
      for (int ni = 0; ni < 2; ++ni)
        acc[mi][ni] = wmma_bf16(af[mi], bfr[ni], acc[mi][ni]);
    __syncthreads();          // done reading `cur` before it is refilled
  }

  const int colB = n0 + wc * 32;
  const int rowB = m0 + wr * 32;
  const int cl   = lane & 15;
  const int rh   = (lane & 16) ? 8 : 0;
#pragma unroll
  for (int mi = 0; mi < 2; ++mi)
#pragma unroll
    for (int ni = 0; ni < 2; ++ni) {
      const int n    = colB + ni * 16 + cl;
      const float bv = bias[n];
      if constexpr (MODE == 3) {
        bf16x8 pack;
#pragma unroll
        for (int j = 0; j < 8; ++j) pack[j] = (bf16_t)(acc[mi][ni][j] + bv);
        *(bf16x8*)((bf16_t*)outp + (size_t)n * M + rowB + mi * 16 + rh) = pack;
      } else {
#pragma unroll
        for (int j = 0; j < 8; ++j) {
          const size_t m   = (size_t)(rowB + mi * 16 + rh + j);
          const size_t idx = m * (size_t)N + n;
          const float  v   = acc[mi][ni][j] + bv;
          if constexpr (MODE == 0) {
            ((bf16_t*)outp)[idx] = (bf16_t)(v * sc);
          } else if constexpr (MODE == 1) {
            ((float*)outp)[idx] = v + Rres[idx];
          } else {
            float g = 0.5f * v * (1.0f + erff(v * 0.70710678118654752f));
            ((bf16_t*)outp)[idx] = (bf16_t)g;
          }
        }
      }
    }
}

// ---------------------------------------------------------------------------
// Flash attention (non-causal), TRANSPOSED-score formulation; zero LDS.
// Per 32-key tile: S^T = K Q^T (4 WMMA), in-lane softmax stats (+1 xor-16
// shuffle each), P^T B-fragment assembled in registers, O^T += V^T P^T
// (4 WMMA, V^T rows = contiguous-key A-fragments).
// Q is pre-scaled by 1/sqrt(dk) in its projection epilogue.
// ---------------------------------------------------------------------------
__global__ __launch_bounds__(128) void flash_attn_kernel(
    const bf16_t* __restrict__ Qs, const bf16_t* __restrict__ Kg,
    const bf16_t* __restrict__ Vt, bf16_t* __restrict__ ctx) {
  const int lane = threadIdx.x & 31;
  const int w    = threadIdx.x >> 5;
  const int b    = blockIdx.y >> 4;
  const int h    = blockIdx.y & 15;
  const int q0   = (blockIdx.x * 4 + w) * 16;
  const size_t rb = (size_t)b * SS;
  const int hoff  = h * DK;
  const int cl = lane & 15;
  const int rh = (lane & 16) ? 8 : 0;
  const bool hiHalf = (lane & 16) != 0;

  const bf16_t* Qp = Qs + (rb + q0) * DD + hoff;
  bf16x16 bq0 = load_frag_bT(Qp,      DD, lane);   // d = 0..31
  bf16x16 bq1 = load_frag_bT(Qp + 32, DD, lane);   // d = 32..63

  f32x8 acc[4] = {};             // O^T tiles: rows d, col = this lane's query
  float mq = -1e30f, lq = 0.f;   // per-lane (= per-query) softmax state

  for (int kv = 0; kv < SS; kv += 32) {
    const bf16_t* Kp = Kg + (rb + kv) * DD + hoff;
    bf16x16 ak00 = load_frag_a(Kp,                DD, lane);
    bf16x16 ak01 = load_frag_a(Kp + 32,           DD, lane);
    bf16x16 ak10 = load_frag_a(Kp + 16 * DD,      DD, lane);
    bf16x16 ak11 = load_frag_a(Kp + 16 * DD + 32, DD, lane);

    if (kv + 32 < SS) {
      __builtin_prefetch(Kp + (size_t)(32 + lane) * DD, 0, 1);
      __builtin_prefetch(Vt + (size_t)(hoff + lane) * MM + rb + kv + 32, 0, 1);
      __builtin_prefetch(Vt + (size_t)(hoff + 32 + lane) * MM + rb + kv + 32, 0, 1);
    }

    f32x8 z = {};
    f32x8 st0 = wmma_bf16(ak00, bq0, z);     // keys 0-15  x queries
    st0       = wmma_bf16(ak01, bq1, st0);
    f32x8 st1 = wmma_bf16(ak10, bq0, z);     // keys 16-31 x queries
    st1       = wmma_bf16(ak11, bq1, st1);

    float tm = fmaxf(st0[0], st1[0]);
#pragma unroll
    for (int j = 1; j < 8; ++j) tm = fmaxf(tm, fmaxf(st0[j], st1[j]));
    tm = fmaxf(tm, __shfl_xor(tm, 16, 32));
    const float nm   = fmaxf(mq, tm);
    const float corr = __expf(mq - nm);
    mq = nm;

    float p0[8], p1[8], ts = 0.f;
#pragma unroll
    for (int j = 0; j < 8; ++j) {
      p0[j] = __expf(st0[j] - nm);
      p1[j] = __expf(st1[j] - nm);
      ts += p0[j] + p1[j];
    }
    ts += __shfl_xor(ts, 16, 32);
    lq = lq * corr + ts;
#pragma unroll
    for (int dt = 0; dt < 4; ++dt)
#pragma unroll
      for (int j = 0; j < 8; ++j) acc[dt][j] *= corr;

    bf16x16 pb;
#pragma unroll
    for (int j = 0; j < 8; ++j) {
      float o0 = __shfl_xor(p0[j], 16, 32);
      float o1 = __shfl_xor(p1[j], 16, 32);
      pb[j]     = (bf16_t)(hiHalf ? o1    : p0[j]);
      pb[j + 8] = (bf16_t)(hiHalf ? p1[j] : o0);
    }

#pragma unroll
    for (int dt = 0; dt < 4; ++dt) {
      bf16x16 av =
          load_frag_a(Vt + (size_t)(hoff + dt * 16) * MM + rb + kv, MM, lane);
      acc[dt] = wmma_bf16(av, pb, acc[dt]);
    }
  }

  const float inv = 1.0f / lq;
  const size_t obase = (rb + q0 + cl) * DD + hoff + rh;
#pragma unroll
  for (int dt = 0; dt < 4; ++dt) {
    bf16x8 pack;
#pragma unroll
    for (int j = 0; j < 8; ++j) pack[j] = (bf16_t)(acc[dt][j] * inv);
    *(bf16x8*)(ctx + obase + dt * 16) = pack;
  }
}

// ---------------------------------------------------------------------------
// LayerNorm over D=1024 (one block per row). Writes f32 and optional bf16.
// ---------------------------------------------------------------------------
__global__ __launch_bounds__(256) void layernorm_kernel(
    const float* __restrict__ in, const float* __restrict__ gamma,
    const float* __restrict__ beta, float* __restrict__ outf,
    bf16_t* __restrict__ outb) {
  __shared__ float red[2][8];
  __shared__ float stats[2];
  const int tid = threadIdx.x, lane = tid & 31, w = tid >> 5;
  const float* p = in + (size_t)blockIdx.x * DD;

  float v[4], s = 0.f, s2 = 0.f;
#pragma unroll
  for (int i = 0; i < 4; ++i) {
    v[i] = p[tid + i * 256];
    s += v[i]; s2 += v[i] * v[i];
  }
#pragma unroll
  for (int m = 1; m < 32; m <<= 1) {
    s  += __shfl_xor(s,  m, 32);
    s2 += __shfl_xor(s2, m, 32);
  }
  if (lane == 0) { red[0][w] = s; red[1][w] = s2; }
  __syncthreads();
  if (tid == 0) {
    float ts = 0.f, ts2 = 0.f;
#pragma unroll
    for (int i = 0; i < 8; ++i) { ts += red[0][i]; ts2 += red[1][i]; }
    stats[0] = ts; stats[1] = ts2;
  }
  __syncthreads();
  const float mu   = stats[0] * (1.0f / DD);
  const float var  = stats[1] * (1.0f / DD) - mu * mu;
  const float rstd = rsqrtf(var + 1e-5f);
#pragma unroll
  for (int i = 0; i < 4; ++i) {
    int c   = tid + i * 256;
    float o = (v[i] - mu) * rstd * gamma[c] + beta[c];
    if (outf) outf[(size_t)blockIdx.x * DD + c] = o;
    if (outb) outb[(size_t)blockIdx.x * DD + c] = (bf16_t)o;
  }
}

// ---------------------------------------------------------------------------
// Workspace layout
// ---------------------------------------------------------------------------
static constexpr size_t B_XD = (size_t)MM * DD * 2;
static constexpr size_t B_W  = (size_t)DD * DD * 2;
static constexpr size_t B_W1 = (size_t)DD * FF * 2;
static constexpr size_t B_H  = (size_t)MM * FF * 2;
static constexpr size_t B_F  = (size_t)MM * DD * 4;

static constexpr size_t OFF_XB  = 0;
static constexpr size_t OFF_WQ  = OFF_XB  + B_XD;
static constexpr size_t OFF_WK  = OFF_WQ  + B_W;
static constexpr size_t OFF_WV  = OFF_WK  + B_W;
static constexpr size_t OFF_WO  = OFF_WV  + B_W;
static constexpr size_t OFF_W1  = OFF_WO  + B_W;
static constexpr size_t OFF_W2  = OFF_W1  + B_W1;
static constexpr size_t OFF_Q   = OFF_W2  + B_W1;
static constexpr size_t OFF_K   = OFF_Q   + B_XD;
static constexpr size_t OFF_VT  = OFF_K   + B_XD;
static constexpr size_t OFF_CTX = OFF_VT  + B_XD;
static constexpr size_t OFF_L1B = OFF_CTX + B_XD;
static constexpr size_t OFF_H   = OFF_L1B + B_XD;
static constexpr size_t OFF_RES = OFF_H   + B_H;
static constexpr size_t OFF_L1F = OFF_RES + B_F;
static constexpr size_t WS_NEED = OFF_L1F + B_F;      // ~248 MiB

extern "C" void kernel_launch(void* const* d_in, const int* in_sizes, int n_in,
                              void* d_out, int out_size, void* d_ws, size_t ws_size,
                              hipStream_t stream) {
  (void)in_sizes; (void)n_in; (void)out_size;
  if (ws_size < WS_NEED) return;

  const float* x    = (const float*)d_in[0];
  const float* w_q  = (const float*)d_in[1];
  const float* b_q  = (const float*)d_in[2];
  const float* w_k  = (const float*)d_in[3];
  const float* b_k  = (const float*)d_in[4];
  const float* w_v  = (const float*)d_in[5];
  const float* b_v  = (const float*)d_in[6];
  const float* w_o  = (const float*)d_in[7];
  const float* b_o  = (const float*)d_in[8];
  const float* w1   = (const float*)d_in[9];
  const float* b1   = (const float*)d_in[10];
  const float* w2   = (const float*)d_in[11];
  const float* b2   = (const float*)d_in[12];
  const float* g1   = (const float*)d_in[13];
  const float* be1  = (const float*)d_in[14];
  const float* g2   = (const float*)d_in[15];
  const float* be2  = (const float*)d_in[16];

  char* ws = (char*)d_ws;
  bf16_t* xb   = (bf16_t*)(ws + OFF_XB);
  bf16_t* wqT  = (bf16_t*)(ws + OFF_WQ);
  bf16_t* wkT  = (bf16_t*)(ws + OFF_WK);
  bf16_t* wvT  = (bf16_t*)(ws + OFF_WV);
  bf16_t* woT  = (bf16_t*)(ws + OFF_WO);
  bf16_t* w1T  = (bf16_t*)(ws + OFF_W1);
  bf16_t* w2T  = (bf16_t*)(ws + OFF_W2);
  bf16_t* Qb   = (bf16_t*)(ws + OFF_Q);
  bf16_t* Kb   = (bf16_t*)(ws + OFF_K);
  bf16_t* VTb  = (bf16_t*)(ws + OFF_VT);
  bf16_t* ctxb = (bf16_t*)(ws + OFF_CTX);
  bf16_t* l1bb = (bf16_t*)(ws + OFF_L1B);
  bf16_t* hb   = (bf16_t*)(ws + OFF_H);
  float*  resf = (float*)(ws + OFF_RES);
  float*  ln1f = (float*)(ws + OFF_L1F);

  const dim3 blk256(256), blk128(128);

  cvt_kernel<<<dim3((MM * DD) / 256), blk256, 0, stream>>>(x, xb, MM * DD);
  cvtT_kernel<<<dim3(DD / 32, DD / 32), blk256, 0, stream>>>(w_q, wqT, DD, DD);
  cvtT_kernel<<<dim3(DD / 32, DD / 32), blk256, 0, stream>>>(w_k, wkT, DD, DD);
  cvtT_kernel<<<dim3(DD / 32, DD / 32), blk256, 0, stream>>>(w_v, wvT, DD, DD);
  cvtT_kernel<<<dim3(DD / 32, DD / 32), blk256, 0, stream>>>(w_o, woT, DD, DD);
  cvtT_kernel<<<dim3(FF / 32, DD / 32), blk256, 0, stream>>>(w1,  w1T, DD, FF);
  cvtT_kernel<<<dim3(DD / 32, FF / 32), blk256, 0, stream>>>(w2,  w2T, FF, DD);

  const dim3 gD(DD / 128, MM / 64);      // (8,128)
  gemm_bf16_wmma<0><<<gD, blk256, 0, stream>>>(xb, wqT, b_q, nullptr, Qb,  MM, DD, DD, 0.125f);
  gemm_bf16_wmma<0><<<gD, blk256, 0, stream>>>(xb, wkT, b_k, nullptr, Kb,  MM, DD, DD, 1.0f);
  gemm_bf16_wmma<3><<<gD, blk256, 0, stream>>>(xb, wvT, b_v, nullptr, VTb, MM, DD, DD, 1.0f);

  flash_attn_kernel<<<dim3(SS / 64, 64), blk128, 0, stream>>>(Qb, Kb, VTb, ctxb);

  gemm_bf16_wmma<1><<<gD, blk256, 0, stream>>>(ctxb, woT, b_o, x, resf, MM, DD, DD, 1.0f);
  layernorm_kernel<<<dim3(MM), blk256, 0, stream>>>(resf, g1, be1, ln1f, l1bb);
  gemm_bf16_wmma<2><<<dim3(FF / 128, MM / 64), blk256, 0, stream>>>(
      l1bb, w1T, b1, nullptr, hb, MM, FF, DD, 1.0f);
  gemm_bf16_wmma<1><<<gD, blk256, 0, stream>>>(hb, w2T, b2, ln1f, resf, MM, DD, FF, 1.0f);
  layernorm_kernel<<<dim3(MM), blk256, 0, stream>>>(resf, g2, be2, (float*)d_out, nullptr);
}
